// FourierLayer_2267742732569
// MI455X (gfx1250) — matrix operand order repaired and checked by reading, not compiled, not run on hardware
//
#include <hip/hip_runtime.h>
#include <hip/hip_bf16.h>
#include <math.h>

// ---------------------------------------------------------------------------
// FourierLayer for MI455X (gfx1250, wave32, WMMA)
//   x: (8, 4096, 512) f32  ->  out: (8, 4096, 512) f32
//
// Pipeline:
//   K0a: build DFT basis A (4096 x 4096): rows 0..2047 = cos(2*pi*k*t/T),
//        rows 2048..4095 = -sin(...); split into f16 hi/lo pair.
//   K0b: transpose/flatten x -> xT (4096 signals x 4096 time), f16 hi/lo.
//   K1 : specT = (A * x)^T via WMMA f16 GEMM with split-f16 error
//        compensation (A_hi*x_hi + A_hi*x_lo + A_lo*x_hi), f32 accum.
//   K2 : per-signal top-7 |X_k|^2 over k=1..2047 + sinusoid reconstruction
//        into outT (signal-major, coalesced).
//   K3 : LDS-tiled transpose outT -> out (b,t,d).
// ---------------------------------------------------------------------------

typedef __attribute__((ext_vector_type(16))) _Float16 v16h;
typedef __attribute__((ext_vector_type(8)))  _Float16 v8h;
typedef __attribute__((ext_vector_type(8)))  float    v8f;

#define T_LEN   4096
#define NSIG    4096
#define TOPK    7
#define W0      1.5339807878856412e-3f   // 2*pi/4096

union Frag16 { v16h v; v8h h[2]; };

// ---------------------------------------------------------------------------
// K0a: basis build (cos rows then -sin rows), split f16 hi/lo
// ---------------------------------------------------------------------------
__global__ __launch_bounds__(256) void build_basis(_Float16* __restrict__ bh,
                                                   _Float16* __restrict__ bl) {
  const long idx = (long)blockIdx.x * 256 + threadIdx.x;   // 16.78M entries
  const int  t   = (int)(idx & 4095);
  const int  m   = (int)(idx >> 12);
  const int  k   = m & 2047;
  const int  p   = (k * t) & 4095;                         // exact integer phase
  float sn, cs;
  __sincosf((float)p * W0, &sn, &cs);
  const float val = (m < 2048) ? cs : -sn;
  const _Float16 h = (_Float16)val;
  bh[idx] = h;
  bl[idx] = (_Float16)(val - (float)h);
}

// ---------------------------------------------------------------------------
// K0b: x (b,t,d) f32 -> xT (n = b*512+d, t) split f16 hi/lo
// ---------------------------------------------------------------------------
__global__ __launch_bounds__(256) void build_xT(const float* __restrict__ x,
                                                _Float16* __restrict__ xh,
                                                _Float16* __restrict__ xl) {
  const long idx = (long)blockIdx.x * 256 + threadIdx.x;   // coalesced read
  const int  d   = (int)(idx & 511);
  const int  t   = (int)((idx >> 9) & 4095);
  const int  b   = (int)(idx >> 21);
  const float v  = x[idx];
  const long  o  = ((long)((b << 9) | d) << 12) + t;
  const _Float16 h = (_Float16)v;
  xh[o] = h;
  xl[o] = (_Float16)(v - (float)h);
}

// ---------------------------------------------------------------------------
// K1: GEMM  spec[m][n] = sum_k A[m][k] * xT[n][k],  stored as specT[n][m].
// 128x128 block tile, 8 waves, each wave owns 2(M) x 4(N) 16x16 WMMA tiles.
// Split-f16: 3 WMMAs per (tile, K-step) -> 24 v_wmma per K-step per wave.
// ---------------------------------------------------------------------------
__global__ __launch_bounds__(256) void dft_gemm(const _Float16* __restrict__ Ah,
                                                const _Float16* __restrict__ Al,
                                                const _Float16* __restrict__ Bh,
                                                const _Float16* __restrict__ Bl,
                                                float* __restrict__ specT) {
  __shared__ _Float16 sAh[128 * 32];
  __shared__ _Float16 sAl[128 * 32];
  __shared__ _Float16 sBh[128 * 32];
  __shared__ _Float16 sBl[128 * 32];

  const int tid   = threadIdx.x;
  const int lane  = tid & 31;
  const int wave  = tid >> 5;
  const int mBase = blockIdx.y * 128;
  const int nBase = blockIdx.x * 128;
  const int wm    = wave >> 1;                 // 4 M-groups (32 rows each)
  const int wn    = wave & 1;                  // 2 N-groups (64 cols each)

  v8f acc[2][4] = {};

  // staging: each thread copies 16 halves (2 x b128) per buffer
  const int  lrow = tid >> 1;                  // 0..127
  const int  lseg = (tid & 1) * 16;            // 0 or 16 (halves)
  const long gA   = (long)(mBase + lrow) * T_LEN;
  const long gB   = (long)(nBase + lrow) * T_LEN;

  // fragment addressing (wave32 WMMA layouts, ISA 7.12.2)
  const int fm    = lane & 15;                 // row within 16x16 subtile
  const int khalf = lane >> 4;                 // K-half selector

  for (int k0 = 0; k0 < T_LEN; k0 += 32) {
    __syncthreads();                           // previous compute done with LDS
    *(v8h*)&sAh[lrow * 32 + lseg]     = *(const v8h*)&Ah[gA + k0 + lseg];
    *(v8h*)&sAh[lrow * 32 + lseg + 8] = *(const v8h*)&Ah[gA + k0 + lseg + 8];
    *(v8h*)&sAl[lrow * 32 + lseg]     = *(const v8h*)&Al[gA + k0 + lseg];
    *(v8h*)&sAl[lrow * 32 + lseg + 8] = *(const v8h*)&Al[gA + k0 + lseg + 8];
    *(v8h*)&sBh[lrow * 32 + lseg]     = *(const v8h*)&Bh[gB + k0 + lseg];
    *(v8h*)&sBh[lrow * 32 + lseg + 8] = *(const v8h*)&Bh[gB + k0 + lseg + 8];
    *(v8h*)&sBl[lrow * 32 + lseg]     = *(const v8h*)&Bl[gB + k0 + lseg];
    *(v8h*)&sBl[lrow * 32 + lseg + 8] = *(const v8h*)&Bl[gB + k0 + lseg + 8];
    if (k0 + 32 < T_LEN) {                     // L2 prefetch of next K panel
      __builtin_prefetch(&Ah[gA + k0 + 32 + lseg], 0, 1);
      __builtin_prefetch(&Bh[gB + k0 + 32 + lseg], 0, 1);
    }
    __syncthreads();

    // A fragments: lane holds row fm; elems 0..7 -> K khalf*8+0..7,
    //              elems 8..15 -> K khalf*8+16..23
    Frag16 ah[2], al[2];
#pragma unroll
    for (int i = 0; i < 2; ++i) {
      const int r = ((wm * 2 + i) * 16 + fm) * 32;
      ah[i].h[0] = *(const v8h*)&sAh[r + khalf * 8];
      ah[i].h[1] = *(const v8h*)&sAh[r + khalf * 8 + 16];
      al[i].h[0] = *(const v8h*)&sAl[r + khalf * 8];
      al[i].h[1] = *(const v8h*)&sAl[r + khalf * 8 + 16];
    }

#pragma unroll
    for (int j = 0; j < 4; ++j) {
      // B fragments: lane holds column fm; elems 0..7 -> K khalf*8+0..7, elems 8..15 -> K khalf*8+16..23
      const int r = ((wn * 4 + j) * 16 + fm) * 32;
      Frag16 bhf, blf;
      bhf.h[0] = *(const v8h*)&sBh[r + khalf * 8];
      bhf.h[1] = *(const v8h*)&sBh[r + khalf * 8 + 16];
      blf.h[0] = *(const v8h*)&sBl[r + khalf * 8];
      blf.h[1] = *(const v8h*)&sBl[r + khalf * 8 + 16];
#pragma unroll
      for (int i = 0; i < 2; ++i) {
        acc[i][j] = __builtin_amdgcn_wmma_f32_16x16x32_f16(
            false, ah[i].v, false, bhf.v, (short)0, acc[i][j], false, false);
        acc[i][j] = __builtin_amdgcn_wmma_f32_16x16x32_f16(
            false, ah[i].v, false, blf.v, (short)0, acc[i][j], false, false);
        acc[i][j] = __builtin_amdgcn_wmma_f32_16x16x32_f16(
            false, al[i].v, false, bhf.v, (short)0, acc[i][j], false, false);
      }
    }
  }

  // epilogue: C/D layout -> lanes 0-15: N=lane, rows 0..7; lanes 16-31: rows 8..15
#pragma unroll
  for (int i = 0; i < 2; ++i)
#pragma unroll
    for (int j = 0; j < 4; ++j) {
      const int n = nBase + (wn * 4 + j) * 16 + fm;
      const int m = mBase + (wm * 2 + i) * 16 + khalf * 8;
      *(v8f*)&specT[(long)n * T_LEN + m] = acc[i][j];
    }
}

// ---------------------------------------------------------------------------
// K2: per-signal top-7 + reconstruction (one workgroup per signal)
// ---------------------------------------------------------------------------
__global__ __launch_bounds__(256) void topk_recon(const float* __restrict__ specT,
                                                  float* __restrict__ outT) {
  __shared__ float amp2[2048];
  __shared__ float rv[256];
  __shared__ int   ri[256];
  __shared__ float cre[TOPK], cim[TOPK];
  __shared__ int   kk[TOPK];

  const int n   = blockIdx.x;
  const int tid = threadIdx.x;
  const float* re = &specT[(long)n * T_LEN];
  const float* im = re + 2048;

  for (int k = tid; k < 2048; k += 256) {
    const float r = re[k], q = im[k];
    amp2[k] = (k == 0) ? -1.0f : (r * r + q * q);   // DC and Nyquist excluded
  }
  __syncthreads();

  for (int it = 0; it < TOPK; ++it) {
    float best = -2.0f;
    int   bi   = 0;
    for (int k = tid; k < 2048; k += 256) {
      const float v = amp2[k];
      if (v > best) { best = v; bi = k; }
    }
    rv[tid] = best;
    ri[tid] = bi;
    __syncthreads();
    for (int s = 128; s > 0; s >>= 1) {
      if (tid < s) {
        const float a = rv[tid], b = rv[tid + s];
        if (b > a || (b == a && ri[tid + s] < ri[tid])) {
          rv[tid] = b;
          ri[tid] = ri[tid + s];
        }
      }
      __syncthreads();
    }
    if (tid == 0) {
      const int kb = ri[0];
      kk[it]  = kb;
      cre[it] = 2.0f * re[kb];   // signal + conjugate -> factor 2
      cim[it] = 2.0f * im[kb];
      amp2[kb] = -3.0f;
    }
    __syncthreads();
  }

  float crl[TOPK], cil[TOPK];
  int   kl[TOPK];
#pragma unroll
  for (int j = 0; j < TOPK; ++j) { crl[j] = cre[j]; cil[j] = cim[j]; kl[j] = kk[j]; }

  for (int t = tid; t < T_LEN; t += 256) {
    float s = 0.0f;
#pragma unroll
    for (int j = 0; j < TOPK; ++j) {
      const int p = (kl[j] * t) & 4095;          // exact phase wrap
      float sn, cs;
      __sincosf(W0 * (float)p, &sn, &cs);
      s += crl[j] * cs - cil[j] * sn;            // 2*(Re*cos - Im*sin)
    }
    outT[(long)n * T_LEN + t] = s;
  }
}

// ---------------------------------------------------------------------------
// K3: outT (n = b*512+d, t) -> out (b, t, d), LDS 32x32 tile transpose
// ---------------------------------------------------------------------------
__global__ __launch_bounds__(256) void transpose_out(const float* __restrict__ outT,
                                                     float* __restrict__ out) {
  __shared__ float tile[32][33];
  const int b  = blockIdx.z;
  const int t0 = blockIdx.x * 32;
  const int d0 = blockIdx.y * 32;
  const int tx = threadIdx.x;     // 0..31
  const int ty = threadIdx.y;     // 0..7
#pragma unroll
  for (int i = 0; i < 4; ++i) {
    const int row = ty + 8 * i;   // d offset
    tile[row][tx] = outT[((long)(b * 512 + d0 + row) << 12) + t0 + tx];
  }
  __syncthreads();
#pragma unroll
  for (int i = 0; i < 4; ++i) {
    const int row = ty + 8 * i;   // t offset
    out[((long)(b * 4096 + t0 + row) << 9) + d0 + tx] = tile[tx][row];
  }
}

// ---------------------------------------------------------------------------
extern "C" void kernel_launch(void* const* d_in, const int* in_sizes, int n_in,
                              void* d_out, int out_size, void* d_ws, size_t ws_size,
                              hipStream_t stream) {
  const float* x   = (const float*)d_in[0];
  float*       out = (float*)d_out;
  char*        ws  = (char*)d_ws;

  const size_t MB = 1024u * 1024u;
  _Float16* basis_hi = (_Float16*)(ws);                  // 32 MB
  _Float16* basis_lo = (_Float16*)(ws + 32 * MB);        // 32 MB
  _Float16* xT_hi    = (_Float16*)(ws + 64 * MB);        // 32 MB
  _Float16* xT_lo    = (_Float16*)(ws + 96 * MB);        // 32 MB
  float*    specT    = (float*)   (ws + 128 * MB);       // 64 MB
  float*    outT     = (float*)   (ws);                  // 64 MB, reuses basis
                                                         // region (dead after K1)

  build_basis<<<dim3(65536), 256, 0, stream>>>(basis_hi, basis_lo);
  build_xT   <<<dim3(65536), 256, 0, stream>>>(x, xT_hi, xT_lo);
  dft_gemm   <<<dim3(32, 32), 256, 0, stream>>>(basis_hi, basis_lo,
                                                xT_hi, xT_lo, specT);
  topk_recon <<<dim3(4096), 256, 0, stream>>>(specT, outT);
  transpose_out<<<dim3(128, 16, 8), dim3(32, 8), 0, stream>>>(outT, out);
}
